// GroupQueryAttention_1202590842990
// MI455X (gfx1250) — matrix operand order repaired and checked by reading, not compiled, not run on hardware
//
#include <hip/hip_runtime.h>
#include <stdint.h>

// ---------------------------------------------------------------------------
// GQA attention layer for gfx1250 (MI455X), bf16 WMMA + async-LDS + TDM.
//   B=2 T=4096 D_MODEL=2048 H=32 KV=8 DH=64 GROUP=4 WINDOW=1024 SINK=4
//   S = SINK + WINDOW = 1028 keys after slicing, padded to SP=1056.
// ---------------------------------------------------------------------------

#define BB      2
#define TT      4096
#define DM      2048
#define NH      32
#define NKV     8
#define DH      64
#define SREAL   1028
#define SPAD    1056
#define ROWS    (BB * TT)          // 8192

typedef __attribute__((ext_vector_type(16))) __bf16 v16bf;
typedef __attribute__((ext_vector_type(8)))  float  v8f;
typedef __attribute__((ext_vector_type(4)))  unsigned int v4u;
typedef __attribute__((ext_vector_type(4)))  unsigned int u32x4;
typedef __attribute__((ext_vector_type(8)))  int i32x8;
typedef __attribute__((ext_vector_type(4)))  int i32x4;

typedef __attribute__((address_space(3))) void AS3V;
typedef __attribute__((address_space(1))) i32x4 GLD128;   // global 16B chunk
typedef __attribute__((address_space(3))) i32x4 LLD128;   // LDS 16B chunk

#if __has_builtin(__builtin_amdgcn_global_load_async_to_lds_b128)
#define HAVE_ASYNC_LDS 1
#else
#define HAVE_ASYNC_LDS 0
#endif

#if __has_builtin(__builtin_amdgcn_tensor_load_to_lds) && \
    __has_builtin(__builtin_amdgcn_s_wait_tensorcnt)
#define HAVE_TDM 1
#else
#define HAVE_TDM 0
#endif

union Frag {
    v4u   u[2];   // two 16-byte halves (8 VGPRs)
    v16bf v;      // 16 bf16 pairs = 8 VGPRs
};

// fp32 -> bf16 (round-to-nearest-even-ish)
static __device__ __forceinline__ unsigned short f2bf(float f) {
    union { float f; unsigned u; } x; x.f = f;
    unsigned r = x.u + 0x7FFFu + ((x.u >> 16) & 1u);
    return (unsigned short)(r >> 16);
}

static __device__ __forceinline__ float rowmax16(float v) {
#pragma unroll
    for (int m = 1; m < 16; m <<= 1) v = fmaxf(v, __shfl_xor(v, m, 16));
    return v;
}
static __device__ __forceinline__ float rowsum16(float v) {
#pragma unroll
    for (int m = 1; m < 16; m <<= 1) v += __shfl_xor(v, m, 16);
    return v;
}

// global (16B) -> LDS: async copy if available (ASYNCcnt path), else ld+st
static __device__ __forceinline__ void copy16_g2l(unsigned short* dst,
                                                  const unsigned short* src) {
#if HAVE_ASYNC_LDS
    __builtin_amdgcn_global_load_async_to_lds_b128((GLD128*)src, (LLD128*)dst, 0, 0);
#else
    *(v4u*)dst = *(const v4u*)src;
#endif
}
static __device__ __forceinline__ void wait_async_le1(bool pending) {
#if HAVE_ASYNC_LDS && __has_builtin(__builtin_amdgcn_s_wait_asynccnt)
    if (pending) __builtin_amdgcn_s_wait_asynccnt(1);
    else         __builtin_amdgcn_s_wait_asynccnt(0);
#else
    (void)pending;
#endif
}

#if HAVE_TDM
// lds byte offset (AS3 numeric address) of a __shared__ object
static __device__ __forceinline__ unsigned lds_off(const void* p) {
    return (unsigned)(unsigned long long)(AS3V*)(void*)p;
}
// TDM 2D tile load: bf16 tile [tile1 rows x tile0 elems], row stride
// stride0 (elements) in global, packed tight in LDS. D# per CDNA5 ISA §8.
static __device__ __forceinline__ void tdm_load_2d(const unsigned short* gp,
                                                   unsigned lds,
                                                   unsigned tile0, unsigned tile1,
                                                   unsigned long long stride0) {
    const unsigned long long ga = (unsigned long long)(const void*)gp;
    u32x4 g0 = { 1u,                                   // count=1, user mode
                 lds,                                  // lds_addr   [63:32]
                 (unsigned)ga,                         // global_addr[95:64]
                 (((unsigned)(ga >> 32)) & 0x01FFFFFFu) | 0x80000000u }; // +type=2
    i32x8 g1;
    g1[0] = (int)(1u << 16);                           // data_size=1 (2B), mask=0
    g1[1] = (int)((tile0 & 0xFFFFu) << 16);            // tensor_dim0[15:0]
    g1[2] = (int)((tile0 >> 16) | ((tile1 & 0xFFFFu) << 16)); // td0 hi | td1 lo
    g1[3] = (int)(((tile1 >> 16) & 0xFFFFu) | (tile0 << 16)); // td1 hi | tile_dim0
    g1[4] = (int)(tile1 & 0xFFFFu);                    // tile_dim1 (tile_dim2=0)
    g1[5] = (int)(unsigned)stride0;                    // dim0_stride[31:0]
    g1[6] = (int)((unsigned)(stride0 >> 32) & 0xFFFFu);// dim0_stride[47:32]
    g1[7] = 0;                                         // dim1_stride unused (2D)
    i32x4 g2 = {0, 0, 0, 0};
    i32x4 g3 = {0, 0, 0, 0};
    i32x8 g4 = {0, 0, 0, 0, 0, 0, 0, 0};               // 6-arg toolchain form
    __builtin_amdgcn_tensor_load_to_lds(g0, g1, g2, g3, g4, 0);
}
#endif

// ---------------------------------------------------------------------------
// fp32 -> bf16 bulk convert
// ---------------------------------------------------------------------------
__global__ void k_f32_to_bf16(const float* __restrict__ in,
                              unsigned short* __restrict__ out, size_t n) {
    size_t i = (size_t)blockIdx.x * blockDim.x + threadIdx.x;
    size_t stride = (size_t)gridDim.x * blockDim.x;
    for (; i < n; i += stride) out[i] = f2bf(in[i]);
}

// ---------------------------------------------------------------------------
// Y[M,N] (fp32) = A[M,K] (bf16) @ W[N,K]^T (bf16).
// Block = 256 thr = 8 waves, block tile 128(M) x 64(N).
// B tile (64x32) staged once per k-step in double-buffered LDS (async copy),
// shared by all 8 waves (8x reuse). Fragment layouts per ISA 7.12.2.
// ---------------------------------------------------------------------------
__global__ void k_gemm_bf16(const unsigned short* __restrict__ A,
                            const unsigned short* __restrict__ W,
                            float* __restrict__ Y, int M, int N, int K) {
    __shared__ unsigned short Bt[2][64 * 32];     // 8 KB double-buffered B tile

    const int lane = threadIdx.x & 31;
    const int wv   = threadIdx.x >> 5;            // wave in block: 0..7
    const int half = lane >> 4;
    const int l16  = lane & 15;

    const int nbt = N >> 6;
    const int mb  = blockIdx.x / nbt;
    const int nb  = blockIdx.x - mb * nbt;
    const int n0  = nb * 64;

    // cooperative B loader: 256 threads x 16 B = 64 rows x 32 k of bf16
    const int brow = threadIdx.x >> 2;
    const int bseg = (threadIdx.x & 3) * 8;
    const unsigned short* Wr = W + (size_t)(n0 + brow) * K + bseg;

    const unsigned short* Ap = A + (size_t)(mb * 128 + wv * 16 + l16) * K;

    v8f acc[4] = {v8f{}, v8f{}, v8f{}, v8f{}};

    copy16_g2l(&Bt[0][brow * 32 + bseg], Wr);     // prologue: k-block 0
    int cur = 0;
    for (int kb = 0; kb < K; kb += 32) {
        const bool has_next = (kb + 32 < K);
        if (has_next) copy16_g2l(&Bt[cur ^ 1][brow * 32 + bseg], Wr + kb + 32);
        wait_async_le1(has_next);                 // own current-buffer load done
        __syncthreads();                          // whole tile visible

        Frag a;
        a.u[0] = *(const v4u*)(Ap + kb + half * 8);
        a.u[1] = *(const v4u*)(Ap + kb + half * 8 + 16);
        if (has_next) __builtin_prefetch(Ap + kb + 32, 0, 3); // global_prefetch_b8

        const unsigned short* Bl = &Bt[cur][0];
#pragma unroll
        for (int j = 0; j < 4; ++j) {
            const unsigned short* Bp = Bl + (j * 16 + l16) * 32 + half * 8;
            Frag b;
            b.u[0] = *(const v4u*)(Bp);           // ds_load_b128
            b.u[1] = *(const v4u*)(Bp + 16);
            acc[j] = __builtin_amdgcn_wmma_f32_16x16x32_bf16(
                false, a.v, false, b.v, (short)0, acc[j], false, false);
        }
        __syncthreads();                          // before overwriting buf
        cur ^= 1;
    }

    // D layout: lane L holds rows (L/16)*8 + r, col L%16
#pragma unroll
    for (int j = 0; j < 4; ++j) {
        const int col = n0 + j * 16 + l16;
#pragma unroll
        for (int r = 0; r < 8; ++r) {
            const int row = mb * 128 + wv * 16 + half * 8 + r;
            Y[(size_t)row * N + col] = acc[j][r];
        }
    }
}

// ---------------------------------------------------------------------------
// RoPE on Q (fp32 in, bf16 out, same [row, h*64+d] layout)
// ---------------------------------------------------------------------------
__global__ void k_rope_q(const float* __restrict__ q,
                         unsigned short* __restrict__ qr) {
    const int tid = blockIdx.x * blockDim.x + threadIdx.x;
    const int d   = tid & 31;
    const int h   = (tid >> 5) & 31;
    const int row = tid >> 10;
    if (row >= ROWS) return;
    const int t = row & (TT - 1);

    const size_t base = (size_t)row * DM + h * DH;
    const float x1 = q[base + d];
    const float x2 = q[base + d + 32];
    const float inv_freq = __powf(10000.0f, -(float)(2 * d) / 64.0f);
    float s, c;
    __sincosf((float)t * inv_freq, &s, &c);
    qr[base + d]      = f2bf(x1 * c - x2 * s);
    qr[base + d + 32] = f2bf(x2 * c + x1 * s);
}

// ---------------------------------------------------------------------------
// K: RoPE + sink/window slice -> Kp[b][hk][s][d] (bf16, SPAD padded)
// V: slice + transpose        -> Vt[b][hk][d][s] (bf16, SPAD padded)
// ---------------------------------------------------------------------------
__global__ void k_kv_prep(const float* __restrict__ kf,
                          const float* __restrict__ vf,
                          unsigned short* __restrict__ Kp,
                          unsigned short* __restrict__ Vt) {
    const int tid = blockIdx.x * blockDim.x + threadIdx.x;
    const int d = tid & 31;
    const int s = (tid >> 5) % SPAD;
    const int h = (tid / (32 * SPAD)) % NKV;
    const int b = tid / (32 * SPAD * NKV);
    if (b >= BB) return;

    const size_t kout  = ((size_t)(b * NKV + h) * SPAD + s) * DH;
    const size_t vbase = (size_t)(b * NKV + h) * DH;

    if (s < SREAL) {
        const int t = (s < 4) ? s : (s + (TT - 1024 - 4));   // sink then window
        const size_t base = (size_t)(b * TT + t) * (NKV * DH) + h * DH;
        const float x1 = kf[base + d];
        const float x2 = kf[base + d + 32];
        const float inv_freq = __powf(10000.0f, -(float)(2 * d) / 64.0f);
        float sn, cs;
        __sincosf((float)t * inv_freq, &sn, &cs);
        Kp[kout + d]      = f2bf(x1 * cs - x2 * sn);
        Kp[kout + d + 32] = f2bf(x2 * cs + x1 * sn);
        Vt[(vbase + d) * SPAD + s]      = f2bf(vf[base + d]);
        Vt[(vbase + d + 32) * SPAD + s] = f2bf(vf[base + d + 32]);
    } else {
        Kp[kout + d]      = 0;
        Kp[kout + d + 32] = 0;
        Vt[(vbase + d) * SPAD + s]      = 0;
        Vt[(vbase + d + 32) * SPAD + s] = 0;
    }
}

// ---------------------------------------------------------------------------
// Flash attention: one wave per (b, h, 16-query tile). 64 thr/block = 2 waves.
// K/V 32-key blocks staged into double-buffered LDS by the Tensor Data Mover
// (TENSORcnt pipeline); P goes D-layout -> A-layout via a per-wave LDS stage.
// ---------------------------------------------------------------------------
#define AT_WAVES 2
__global__ void k_attn(const unsigned short* __restrict__ qr,
                       const unsigned short* __restrict__ Kp,
                       const unsigned short* __restrict__ Vt,
                       unsigned short* __restrict__ ctx) {
    __shared__ unsigned short sP[AT_WAVES][16 * 32];       // 2 KB
#if HAVE_TDM
    __shared__ unsigned short sK[AT_WAVES][2][32 * DH];    // 16 KB
    __shared__ unsigned short sV[AT_WAVES][2][DH * 32];    // 16 KB
#endif

    const int gwid = (int)((blockIdx.x * blockDim.x + threadIdx.x) >> 5);
    const int lw   = (threadIdx.x >> 5) % AT_WAVES;
    const int lane = threadIdx.x & 31;
    const int half = lane >> 4;
    const int l16  = lane & 15;

    const int qt = gwid & 255;            // T/16 query tiles
    const int h  = (gwid >> 8) & 31;
    const int b  = gwid >> 13;
    if (b >= BB) return;

    const int qbase = qt * 16;
    const int hk = h >> 2;                // GQA group of 4
    const unsigned short* Kh = Kp + (size_t)(b * NKV + hk) * SPAD * DH;
    const unsigned short* Vh = Vt + (size_t)(b * NKV + hk) * DH * SPAD;

    // Q A-fragments (d = 0..31, 32..63)
    const unsigned short* Qp = qr + (size_t)(b * TT + qbase + l16) * DM + h * DH;
    Frag aq[2];
#pragma unroll
    for (int dk = 0; dk < 2; ++dk) {
        aq[dk].u[0] = *(const v4u*)(Qp + dk * 32 + half * 8);
        aq[dk].u[1] = *(const v4u*)(Qp + dk * 32 + half * 8 + 16);
    }

    float mrow[8], lrow[8];
#pragma unroll
    for (int r = 0; r < 8; ++r) { mrow[r] = -1e30f; lrow[r] = 0.0f; }
    v8f acc[4] = {v8f{}, v8f{}, v8f{}, v8f{}};

    unsigned short* P = sP[lw];
    const float scale = 0.125f;           // 1/sqrt(64)

    const int active = min(SREAL, qbase + 16);   // top-left causal: s <= qi
    const int nkb = (active + 31) / 32;

#if HAVE_TDM
    // prologue: stage key-block 0 (K: 32x64 rows-of-d; V: 64x32 rows-of-keys)
    tdm_load_2d(Kh, lds_off(&sK[lw][0][0]), DH, 32, DH);
    tdm_load_2d(Vh, lds_off(&sV[lw][0][0]), 32, DH, SPAD);
#endif
    int cur = 0;

    for (int kb = 0; kb < nkb; ++kb) {
        const int k0 = kb * 32;

#if HAVE_TDM
        if (kb + 1 < nkb) {               // prefetch next block, then wait cur
            tdm_load_2d(Kh + (size_t)(k0 + 32) * DH,
                        lds_off(&sK[lw][cur ^ 1][0]), DH, 32, DH);
            tdm_load_2d(Vh + (k0 + 32),
                        lds_off(&sV[lw][cur ^ 1][0]), 32, DH, SPAD);
            __builtin_amdgcn_s_wait_tensorcnt(2);
        } else {
            __builtin_amdgcn_s_wait_tensorcnt(0);
        }
        asm volatile("" ::: "memory");    // TDM wrote LDS behind compiler's back
        const unsigned short* Kbase = &sK[lw][cur][0];   // [key][64]
        const unsigned short* Vbase = &sV[lw][cur][0];   // [d][32]
        const int VS = 32;
#else
        const unsigned short* Kbase = Kh + (size_t)k0 * DH;
        const unsigned short* Vbase = Vh + k0;
        const int VS = SPAD;
#endif

        // ---- scores: two 16x16 tiles (32 keys), reduce over d via 2 WMMAs
        v8f sc[2];
#pragma unroll
        for (int sub = 0; sub < 2; ++sub) {
            const unsigned short* Kb = Kbase + (size_t)(sub * 16 + l16) * DH;
            v8f c = {};
#pragma unroll
            for (int dk = 0; dk < 2; ++dk) {
                Frag bk;
                bk.u[0] = *(const v4u*)(Kb + dk * 32 + half * 8);
                bk.u[1] = *(const v4u*)(Kb + dk * 32 + half * 8 + 16);
                c = __builtin_amdgcn_wmma_f32_16x16x32_bf16(
                    false, aq[dk].v, false, bk.v, (short)0, c, false, false);
            }
            sc[sub] = c;
        }

        // ---- online softmax over this 32-key block
#pragma unroll
        for (int r = 0; r < 8; ++r) {
            const int qi = qbase + half * 8 + r;
            const int s0 = k0 + l16;
            const int s1 = k0 + 16 + l16;
            float v0 = (s0 <= qi && s0 < SREAL) ? sc[0][r] * scale : -1e30f;
            float v1 = (s1 <= qi && s1 < SREAL) ? sc[1][r] * scale : -1e30f;
            float mb = rowmax16(fmaxf(v0, v1));
            float mn = fmaxf(mrow[r], mb);
            float al = __expf(mrow[r] - mn);
            mrow[r] = mn;
            float p0 = __expf(v0 - mn);
            float p1 = __expf(v1 - mn);
            lrow[r] = lrow[r] * al + rowsum16(p0 + p1);
            acc[0][r] *= al; acc[1][r] *= al; acc[2][r] *= al; acc[3][r] *= al;
            P[(half * 8 + r) * 32 + l16]      = f2bf(p0);   // D -> row-major
            P[(half * 8 + r) * 32 + 16 + l16] = f2bf(p1);
        }
        asm volatile("" ::: "memory");
        __builtin_amdgcn_wave_barrier();

        // ---- read P back as an A-fragment (16 queries x 32 keys)
        Frag pa;
        const unsigned short* Pl = P + l16 * 32;
        pa.u[0] = *(const v4u*)(Pl + half * 8);
        pa.u[1] = *(const v4u*)(Pl + half * 8 + 16);

        // ---- ctx += P @ V
#pragma unroll
        for (int j = 0; j < 4; ++j) {
            const unsigned short* Vb = Vbase + (size_t)(j * 16 + l16) * VS + half * 8;
            Frag bv;
            bv.u[0] = *(const v4u*)(Vb);
            bv.u[1] = *(const v4u*)(Vb + 16);
            acc[j] = __builtin_amdgcn_wmma_f32_16x16x32_bf16(
                false, pa.v, false, bv.v, (short)0, acc[j], false, false);
        }
        asm volatile("" ::: "memory");
        __builtin_amdgcn_wave_barrier();
        cur ^= 1;
    }

    // ---- epilogue: normalize, store bf16 ctx tile [16 x 64]
#pragma unroll
    for (int r = 0; r < 8; ++r) {
        const float inv = 1.0f / lrow[r];
        const size_t rowoff = (size_t)(b * TT + qbase + half * 8 + r) * DM + h * DH;
#pragma unroll
        for (int j = 0; j < 4; ++j)
            ctx[rowoff + j * 16 + l16] = f2bf(acc[j][r] * inv);
    }
}

// ---------------------------------------------------------------------------
// Host-side launch
// ---------------------------------------------------------------------------
extern "C" void kernel_launch(void* const* d_in, const int* in_sizes, int n_in,
                              void* d_out, int out_size, void* d_ws, size_t ws_size,
                              hipStream_t stream) {
    const float* x  = (const float*)d_in[0];   // [8192, 2048]
    const float* wq = (const float*)d_in[1];   // [2048, 2048]
    const float* wk = (const float*)d_in[2];   // [512, 2048]
    const float* wv = (const float*)d_in[3];   // [512, 2048]
    const float* wp = (const float*)d_in[4];   // [2048, 2048]
    float* out = (float*)d_out;                // [8192, 2048]

    char* ws = (char*)d_ws;
    size_t off = 0;
    auto alloc = [&](size_t bytes) {
        void* p = ws + off; off += (bytes + 255) & ~(size_t)255; return p;
    };
    unsigned short* xb  = (unsigned short*)alloc((size_t)ROWS * DM * 2);
    unsigned short* wqb = (unsigned short*)alloc((size_t)DM * DM * 2);
    unsigned short* wkb = (unsigned short*)alloc((size_t)(NKV * DH) * DM * 2);
    unsigned short* wvb = (unsigned short*)alloc((size_t)(NKV * DH) * DM * 2);
    unsigned short* wpb = (unsigned short*)alloc((size_t)DM * DM * 2);
    float*          qf  = (float*)alloc((size_t)ROWS * DM * 4);
    float*          kf  = (float*)alloc((size_t)ROWS * (NKV * DH) * 4);
    float*          vf  = (float*)alloc((size_t)ROWS * (NKV * DH) * 4);
    unsigned short* qrp = (unsigned short*)alloc((size_t)ROWS * DM * 2);
    unsigned short* Kp  = (unsigned short*)alloc((size_t)BB * NKV * SPAD * DH * 2);
    unsigned short* Vt  = (unsigned short*)alloc((size_t)BB * NKV * DH * SPAD * 2);
    unsigned short* ctx = (unsigned short*)alloc((size_t)ROWS * DM * 2);
    (void)ws_size; (void)n_in; (void)in_sizes; (void)out_size;

    // 1) fp32 -> bf16
    k_f32_to_bf16<<<4096, 256, 0, stream>>>(x,  xb,  (size_t)ROWS * DM);
    k_f32_to_bf16<<<2048, 256, 0, stream>>>(wq, wqb, (size_t)DM * DM);
    k_f32_to_bf16<<<1024, 256, 0, stream>>>(wk, wkb, (size_t)(NKV * DH) * DM);
    k_f32_to_bf16<<<1024, 256, 0, stream>>>(wv, wvb, (size_t)(NKV * DH) * DM);
    k_f32_to_bf16<<<2048, 256, 0, stream>>>(wp, wpb, (size_t)DM * DM);

    // 2) QKV projections (block tile 128x64, 8 waves/block)
    k_gemm_bf16<<<(ROWS / 128) * (DM / 64), 256, 0, stream>>>(xb, wqb, qf, ROWS, DM, DM);
    k_gemm_bf16<<<(ROWS / 128) * ((NKV * DH) / 64), 256, 0, stream>>>(xb, wkb, kf, ROWS, NKV * DH, DM);
    k_gemm_bf16<<<(ROWS / 128) * ((NKV * DH) / 64), 256, 0, stream>>>(xb, wvb, vf, ROWS, NKV * DH, DM);

    // 3) RoPE(Q); RoPE+slice(K), slice+transpose(V)
    k_rope_q<<<(ROWS * NH * 32) / 256, 256, 0, stream>>>(qf, qrp);
    k_kv_prep<<<(BB * NKV * SPAD * 32) / 256, 256, 0, stream>>>(kf, vf, Kp, Vt);

    // 4) flash attention (2 waves/block, TDM-staged K/V)
    k_attn<<<(BB * NH * (TT / 16)) / AT_WAVES, AT_WAVES * 32, 0, stream>>>(qrp, Kp, Vt, ctx);

    // 5) output projection -> fp32 d_out
    k_gemm_bf16<<<(ROWS / 128) * (DM / 64), 256, 0, stream>>>(ctx, wpb, out, ROWS, DM, DM);
}
